// HAttnEncoderLayer_1511828488806
// MI455X (gfx1250) — matrix operand order repaired and checked by reading, not compiled, not run on hardware
//
#include <hip/hip_runtime.h>

// ---------------------------------------------------------------------------
// Problem constants
// ---------------------------------------------------------------------------
#define NB      16
#define LSEQ    1024
#define EDIM    512
#define NHEAD   8
#define HD      64
#define DFF     2048
#define NLEV    10
#define NSCALE  11

typedef __attribute__((ext_vector_type(16))) __bf16 v16bf;
typedef __attribute__((ext_vector_type(8)))  __bf16 v8bf;
typedef __attribute__((ext_vector_type(8)))  float  v8f;
typedef __attribute__((ext_vector_type(4)))  unsigned int u32x4;
typedef __attribute__((ext_vector_type(8)))  int i32x8;
typedef __attribute__((ext_vector_type(4)))  int i32x4;

__device__ __forceinline__ __bf16 f2bf(float f) {
  unsigned u = __builtin_bit_cast(unsigned, f);
  unsigned r = (u + 0x7FFFu + ((u >> 16) & 1u)) >> 16;   // round-to-nearest-even
  return __builtin_bit_cast(__bf16, (unsigned short)r);
}

// ---------------------------------------------------------------------------
// TDM: async 2-D bf16 tile load (global -> LDS) with padded LDS rows.
// Tile is 32 elements wide (64 B rows) x tile_rows tall; LDS rows padded by
// 16 B (pad_interval = 16 DWORDs -> code 3, pad_amount = 4 DWORDs -> code 3)
// to match the 40-element LDS row stride used by the WMMA fragment reads.
// Rows beyond tensor_rows_rem are zero-filled by TDM OOB semantics.
// ---------------------------------------------------------------------------
__device__ __forceinline__ void tdm_load_tile(
    unsigned lds_addr, unsigned long long gaddr,
    int tensor_cols_rem, int row_stride_elems, int tensor_rows_rem,
    int tile_rows) {
  u32x4 g0;
  g0[0] = 1u;                                             // count=1 (valid), user mode
  g0[1] = lds_addr;                                       // lds_addr [63:32]
  g0[2] = (unsigned)(gaddr & 0xFFFFFFFFull);              // global_addr[31:0]
  g0[3] = (unsigned)((gaddr >> 32) & 0x1FFFFFFull)        // global_addr[56:32]
          | (2u << 30);                                   // type = 2 ("image")
  i32x8 g1;
  g1[0] = (int)((1u << 16)      // data_size = 2 bytes
              | (1u << 20)      // pad_enable
              | (3u << 22)      // pad_interval: 16 DWORDs (64 B) per row
              | (3u << 25));    // pad_amount: 4 DWORDs (16 B)
  g1[1] = (int)(((unsigned)tensor_cols_rem & 0xFFFFu) << 16);        // dim0[15:0]
  g1[2] = (int)((((unsigned)tensor_cols_rem >> 16) & 0xFFFFu)
              | (((unsigned)tensor_rows_rem & 0xFFFFu) << 16));      // dim0 hi | dim1 lo
  g1[3] = (int)((((unsigned)tensor_rows_rem >> 16) & 0xFFFFu)
              | (32u << 16));                                        // dim1 hi | tile_dim0=32
  g1[4] = (int)((unsigned)tile_rows & 0xFFFFu);                      // tile_dim1 | tile_dim2=0
  g1[5] = (int)(unsigned)row_stride_elems;                           // dim0_stride[31:0]
  g1[6] = 0;                                                         // stride hi | dim1_stride
  g1[7] = 0;
  i32x4 z4 = {0, 0, 0, 0};
#if defined(__clang_major__) && (__clang_major__ >= 23)
  i32x8 z8 = {0, 0, 0, 0, 0, 0, 0, 0};
  __builtin_amdgcn_tensor_load_to_lds(g0, g1, z4, z4, z8, 0);
#else
  __builtin_amdgcn_tensor_load_to_lds(g0, g1, z4, z4, 0);
#endif
}

// ---------------------------------------------------------------------------
// fp32 -> bf16 conversion
// ---------------------------------------------------------------------------
__global__ __launch_bounds__(256) void f32_to_bf16_kernel(
    const float* __restrict__ in, __bf16* __restrict__ out, int n) {
  int i = blockIdx.x * 256 + threadIdx.x;
  if (i < n) out[i] = f2bf(in[i]);
}

// ---------------------------------------------------------------------------
// One conv-pyramid level (token-major). Emits fp32 (next level) + bf16 (GEMMs).
// ---------------------------------------------------------------------------
__global__ __launch_bounds__(256) void conv_level_kernel(
    const float* __restrict__ in, const float* __restrict__ W,
    const float* __restrict__ bias, float* __restrict__ out,
    __bf16* __restrict__ outbf, int l_out) {
  int idx = blockIdx.x * 256 + threadIdx.x;
  int total = NB * l_out * EDIM;
  if (idx >= total) return;
  int e  = idx & (EDIM - 1);
  int t  = (idx >> 9) % l_out;
  int n  = idx / (EDIM * l_out);
  int co = e & (HD - 1);
  int hb = e & ~(HD - 1);
  int l_in = l_out * 2;
  const float* i0 = in + ((size_t)n * l_in + 2 * t) * EDIM + hb;
  const float* i1 = i0 + EDIM;
  const float* wr = W + co * (HD * 2);
  float acc = bias[co];
#pragma unroll 8
  for (int ci = 0; ci < HD; ++ci)
    acc += i0[ci] * wr[2 * ci] + i1[ci] * wr[2 * ci + 1];
  float r = fmaxf(acc, 0.0f);
  out[idx]   = r;
  outbf[idx] = f2bf(r);
}

// ---------------------------------------------------------------------------
// TDM-fed bf16 WMMA GEMM:  C[M x Nc] = act( A[M x K] * W^T + bias )
//   A: bf16 row-major.  W: bf16 (Nc x K) row-major (PyTorch layout).
// Block tile 128x64, 8 waves as 4(m) x 2(n); each wave owns a 32x32 strip =
// 4 accumulators -> 4 v_wmma_f32_16x16x32_bf16 per 32-wide K-step.
// A/W tiles arrive via tensor_load_to_lds (wave0 / wave1), double-buffered.
// ---------------------------------------------------------------------------
#define LDSROW 40

union FragBF { v16bf v; v8bf h[2]; };

__global__ __launch_bounds__(256) void gemm_tdm_bf16_kernel(
    const __bf16* __restrict__ Ab, const __bf16* __restrict__ Wb,
    const float* __restrict__ bias, float* __restrict__ Cf,
    __bf16* __restrict__ Cbf, int M, int Nc, int K, int relu) {
  __shared__ __bf16 Ats[2][128 * LDSROW];
  __shared__ __bf16 Wts[2][64 * LDSROW];

  const int tid  = threadIdx.x;
  const int lane = tid & 31;
  const int wv   = tid >> 5;
  const int m_w  = wv >> 1;      // 0..3 -> 32-row strip
  const int n_w  = wv & 1;       // 0..1 -> 32-col strip
  const int hf   = lane >> 4;
  const int l16  = lane & 15;

  const int block_m = blockIdx.y * 128;
  const int block_n = blockIdx.x * 64;
  const int rows_rem = M - block_m;      // TDM zero-fills rows beyond this

  const unsigned long long abase =
      (unsigned long long)(uintptr_t)(Ab + (size_t)block_m * K);
  const unsigned long long wbase =
      (unsigned long long)(uintptr_t)(Wb + (size_t)block_n * K);
  const unsigned a_lds0 = (unsigned)(uintptr_t)(void*)&Ats[0][0];
  const unsigned a_lds1 = (unsigned)(uintptr_t)(void*)&Ats[1][0];
  const unsigned w_lds0 = (unsigned)(uintptr_t)(void*)&Wts[0][0];
  const unsigned w_lds1 = (unsigned)(uintptr_t)(void*)&Wts[1][0];

  // prologue: fetch K-step 0 into buffer 0
  if (wv == 0) tdm_load_tile(a_lds0, abase, K, K, rows_rem, 128);
  if (wv == 1) tdm_load_tile(w_lds0, wbase, K, K, 64, 64);
  if (wv < 2) __builtin_amdgcn_s_wait_tensorcnt(0);
  __syncthreads();

  v8f acc[2][2] = {};
  const int nk = K >> 5;
  for (int i = 0; i < nk; ++i) {
    const int cur = i & 1;
    // prefetch next K-step into the other buffer while we compute
    if (i + 1 < nk) {
      const int k0n = (i + 1) << 5;
      if (wv == 0)
        tdm_load_tile(cur ? a_lds0 : a_lds1, abase + 2ull * k0n,
                      K - k0n, K, rows_rem, 128);
      if (wv == 1)
        tdm_load_tile(cur ? w_lds0 : w_lds1, wbase + 2ull * k0n,
                      K - k0n, K, 64, 64);
    }

    // contiguous-k fragments (k = 16*hf + s, same order for A and B)
    FragBF af[2], bfr[2];
#pragma unroll
    for (int mi = 0; mi < 2; ++mi) {
      const __bf16* ar = &Ats[cur][(m_w * 32 + mi * 16 + l16) * LDSROW];
      af[mi].h[0] = *(const v8bf*)(ar + 16 * hf);
      af[mi].h[1] = *(const v8bf*)(ar + 16 * hf + 8);
    }
#pragma unroll
    for (int ni = 0; ni < 2; ++ni) {
      const __bf16* br = &Wts[cur][(n_w * 32 + ni * 16 + l16) * LDSROW + 16 * hf];
      bfr[ni].h[0] = *(const v8bf*)br;
      bfr[ni].h[1] = *(const v8bf*)(br + 8);
    }
#pragma unroll
    for (int mi = 0; mi < 2; ++mi)
#pragma unroll
      for (int ni = 0; ni < 2; ++ni)
        acc[mi][ni] = __builtin_amdgcn_wmma_f32_16x16x32_bf16(
            false, af[mi].v, false, bfr[ni].v, (short)0, acc[mi][ni],
            false, false);

    if (wv < 2) __builtin_amdgcn_s_wait_tensorcnt(0);
    __syncthreads();
  }

  // epilogue: bias (+ReLU) then fp32 or bf16 store
#pragma unroll
  for (int ni = 0; ni < 2; ++ni) {
    const int col = block_n + n_w * 32 + ni * 16 + l16;
    const float bv = bias[col];
#pragma unroll
    for (int mi = 0; mi < 2; ++mi) {
      const int row0 = block_m + m_w * 32 + mi * 16 + 8 * hf;
#pragma unroll
      for (int r = 0; r < 8; ++r) {
        const int gr = row0 + r;
        if (gr < M) {
          float v = acc[mi][ni][r] + bv;
          if (relu) v = fmaxf(v, 0.0f);
          if (Cbf) Cbf[(size_t)gr * Nc + col] = f2bf(v);
          else     Cf [(size_t)gr * Nc + col] = v;
        }
      }
    }
  }
}

// ---------------------------------------------------------------------------
// Attention: one wave per (token, head); 11 keys (one per scale), coarse K/V
// looked up at t >> s (== the reference's repeat_interleave). bf16 output
// (only consumed by the out-proj GEMM).
// ---------------------------------------------------------------------------
__global__ __launch_bounds__(256) void attn_kernel(
    const float* __restrict__ Q, const float* __restrict__ Kb,
    const float* __restrict__ Vb, __bf16* __restrict__ Obf) {
  int unit = blockIdx.x * 8 + (threadIdx.x >> 5);
  int lane = threadIdx.x & 31;
  int b = unit >> 3;
  int h = unit & 7;
  int n = b >> 10;
  int t = b & (LSEQ - 1);
  int d0 = lane * 2;

  size_t qoff = (size_t)b * EDIM + h * HD + d0;
  float2 qv = *(const float2*)(Q + qoff);

  float sc[NSCALE];
  float2 vv[NSCALE];
  size_t soff = 0;
#pragma unroll
  for (int s = 0; s < NSCALE; ++s) {
    int ls = LSEQ >> s;
    size_t base = soff + ((size_t)(n * ls + (t >> s))) * EDIM + h * HD + d0;
    float2 kv = *(const float2*)(Kb + base);
    vv[s] = *(const float2*)(Vb + base);
    float p = qv.x * kv.x + qv.y * kv.y;
    p += __shfl_xor(p, 16, 32);
    p += __shfl_xor(p, 8, 32);
    p += __shfl_xor(p, 4, 32);
    p += __shfl_xor(p, 2, 32);
    p += __shfl_xor(p, 1, 32);
    sc[s] = p * 0.125f;                 // 1/sqrt(64)
    soff += (size_t)(NB * EDIM) * ls;
  }
  float mx = sc[0];
#pragma unroll
  for (int s = 1; s < NSCALE; ++s) mx = fmaxf(mx, sc[s]);
  float sum = 0.0f;
#pragma unroll
  for (int s = 0; s < NSCALE; ++s) { sc[s] = __expf(sc[s] - mx); sum += sc[s]; }
  float inv = 1.0f / sum;
  float o0 = 0.0f, o1 = 0.0f;
#pragma unroll
  for (int s = 0; s < NSCALE; ++s) {
    float p = sc[s] * inv;
    o0 += p * vv[s].x;
    o1 += p * vv[s].y;
  }
  unsigned pk = (unsigned)__builtin_bit_cast(unsigned short, f2bf(o0))
              | ((unsigned)__builtin_bit_cast(unsigned short, f2bf(o1)) << 16);
  *(unsigned*)(Obf + qoff) = pk;
}

// ---------------------------------------------------------------------------
// LayerNorm (512 cols), optional residual, optional extra bf16 output.
// ---------------------------------------------------------------------------
__global__ __launch_bounds__(256) void ln_kernel(
    const float* __restrict__ in, const float* __restrict__ res,
    const float* __restrict__ w, const float* __restrict__ bias,
    float* __restrict__ out, __bf16* __restrict__ outbf, int rows) {
  int row  = blockIdx.x * 8 + (threadIdx.x >> 5);
  int lane = threadIdx.x & 31;
  if (row >= rows) return;
  const float* p  = in + (size_t)row * EDIM;
  const float* rp = res ? res + (size_t)row * EDIM : nullptr;
  float v[16];
  float sum = 0.0f;
#pragma unroll
  for (int i = 0; i < 16; ++i) {
    int c = lane + 32 * i;
    float x = p[c];
    if (rp) x += rp[c];
    v[i] = x;
    sum += x;
  }
  sum += __shfl_xor(sum, 16, 32); sum += __shfl_xor(sum, 8, 32);
  sum += __shfl_xor(sum, 4, 32);  sum += __shfl_xor(sum, 2, 32);
  sum += __shfl_xor(sum, 1, 32);
  float mean = sum * (1.0f / EDIM);
  float var = 0.0f;
#pragma unroll
  for (int i = 0; i < 16; ++i) { float d = v[i] - mean; var += d * d; }
  var += __shfl_xor(var, 16, 32); var += __shfl_xor(var, 8, 32);
  var += __shfl_xor(var, 4, 32);  var += __shfl_xor(var, 2, 32);
  var += __shfl_xor(var, 1, 32);
  float invs = rsqrtf(var * (1.0f / EDIM) + 1e-5f);
  float* o = out + (size_t)row * EDIM;
#pragma unroll
  for (int i = 0; i < 16; ++i) {
    int c = lane + 32 * i;
    float y = (v[i] - mean) * invs * w[c] + bias[c];
    o[c] = y;
    if (outbf) outbf[(size_t)row * EDIM + c] = f2bf(y);
  }
}

// ---------------------------------------------------------------------------
// Host orchestration
// ---------------------------------------------------------------------------
extern "C" void kernel_launch(void* const* d_in, const int* in_sizes, int n_in,
                              void* d_out, int out_size, void* d_ws, size_t ws_size,
                              hipStream_t stream) {
  (void)in_sizes; (void)n_in; (void)out_size; (void)ws_size;

  const float* src        = (const float*)d_in[0];
  const float* conv_w     = (const float*)d_in[1];
  const float* conv_b     = (const float*)d_in[2];
  const float* in_proj_w  = (const float*)d_in[3];
  const float* in_proj_b  = (const float*)d_in[4];
  const float* out_proj_w = (const float*)d_in[5];
  const float* out_proj_b = (const float*)d_in[6];
  const float* ln1_w      = (const float*)d_in[7];
  const float* ln1_b      = (const float*)d_in[8];
  const float* ln2_w      = (const float*)d_in[9];
  const float* ln2_b      = (const float*)d_in[10];
  const float* w1         = (const float*)d_in[11];
  const float* b1         = (const float*)d_in[12];
  const float* w2         = (const float*)d_in[13];
  const float* b2         = (const float*)d_in[14];

  const int M = NB * LSEQ;                              // 16384
  const size_t HID_ELEMS = (size_t)NB * EDIM * 1023;    // scales 1..10
  const size_t ME        = (size_t)M * EDIM;
  const size_t KV_ELEMS  = (size_t)NB * EDIM * 2047;    // scales 0..10

  float* fp   = (float*)d_ws;
  float* HID  = fp;                       // fp32 hiddens (conv chain)
  float* Qb   = HID + HID_ELEMS;
  float* Kb   = Qb + ME;
  float* Vb   = Kb + KV_ELEMS;
  float* Ob   = Kb + 2 * KV_ELEMS;        // FFN2 output (fp32)
  float* S2   = Ob + ME;
  float* Xb   = S2 + ME;
  __bf16* SRCbf = (__bf16*)(Xb + ME);
  __bf16* HIDbf = SRCbf + ME;
  __bf16* Obf   = HIDbf + HID_ELEMS;      // attention output (bf16)
  __bf16* Xbf   = Obf + ME;
  __bf16* WBF   = Xbf + ME;
  __bf16* Wq = WBF;
  __bf16* Wk = WBF + 512 * 512;
  __bf16* Wv = WBF + 2 * 512 * 512;
  __bf16* Wo = WBF + 3 * 512 * 512;
  __bf16* W1 = Wo + 512 * 512;
  __bf16* W2 = W1 + (size_t)DFF * EDIM;
  __bf16* FFHbf = (__bf16*)Kb;            // reuse K/V region (disjoint lifetime)

  // 1) bf16 conversions: weights + src
  f32_to_bf16_kernel<<<(3 * 512 * 512 + 255) / 256, 256, 0, stream>>>(in_proj_w, WBF, 3 * 512 * 512);
  f32_to_bf16_kernel<<<(512 * 512 + 255) / 256, 256, 0, stream>>>(out_proj_w, Wo, 512 * 512);
  f32_to_bf16_kernel<<<(DFF * EDIM + 255) / 256, 256, 0, stream>>>(w1, W1, DFF * EDIM);
  f32_to_bf16_kernel<<<(EDIM * DFF + 255) / 256, 256, 0, stream>>>(w2, W2, EDIM * DFF);
  f32_to_bf16_kernel<<<((int)ME + 255) / 256, 256, 0, stream>>>(src, SRCbf, (int)ME);

  size_t hoff[12];
  hoff[1] = 0;
  for (int s = 2; s <= 11; ++s)
    hoff[s] = hoff[s - 1] + (size_t)NB * EDIM * (LSEQ >> (s - 1));

  // 2) conv pyramid
  for (int lev = 0; lev < NLEV; ++lev) {
    int l_out = LSEQ >> (lev + 1);
    const float* inp = (lev == 0) ? src : HID + hoff[lev];
    int total = NB * l_out * EDIM;
    conv_level_kernel<<<(total + 255) / 256, 256, 0, stream>>>(
        inp, conv_w + (size_t)lev * HD * HD * 2, conv_b + lev * HD,
        HID + hoff[lev + 1], HIDbf + hoff[lev + 1], l_out);
  }

  // 3) Q projection
  gemm_tdm_bf16_kernel<<<dim3(EDIM / 64, M / 128), 256, 0, stream>>>(
      SRCbf, Wq, in_proj_b, Qb, nullptr, M, EDIM, EDIM, 0);

  // 4) K/V projections on coarse scales
  {
    size_t koff = 0;
    for (int s = 0; s <= 10; ++s) {
      int ls = LSEQ >> s;
      int Ms = NB * ls;
      const __bf16* Ain = (s == 0) ? SRCbf : HIDbf + hoff[s];
      dim3 g(EDIM / 64, (Ms + 127) / 128);
      gemm_tdm_bf16_kernel<<<g, 256, 0, stream>>>(
          Ain, Wk, in_proj_b + EDIM, Kb + koff, nullptr, Ms, EDIM, EDIM, 0);
      gemm_tdm_bf16_kernel<<<g, 256, 0, stream>>>(
          Ain, Wv, in_proj_b + 2 * EDIM, Vb + koff, nullptr, Ms, EDIM, EDIM, 0);
      koff += (size_t)Ms * EDIM;
    }
  }

  // 5) attention over 11 scales -> bf16
  attn_kernel<<<M * NHEAD / 8, 256, 0, stream>>>(Qb, Kb, Vb, Obf);

  // 6) output projection -> src2 (fp32)
  gemm_tdm_bf16_kernel<<<dim3(EDIM / 64, M / 128), 256, 0, stream>>>(
      Obf, Wo, out_proj_b, S2, nullptr, M, EDIM, EDIM, 0);

  // 7) LN1 (reference overwrites residual with norm1(src2))
  ln_kernel<<<M / 8, 256, 0, stream>>>(S2, nullptr, ln1_w, ln1_b, Xb, Xbf, M);

  // 8) FFN: ff1 (bf16 out + ReLU), ff2 (fp32 out)
  gemm_tdm_bf16_kernel<<<dim3(DFF / 64, M / 128), 256, 0, stream>>>(
      Xbf, W1, b1, nullptr, FFHbf, M, DFF, EDIM, 1);
  gemm_tdm_bf16_kernel<<<dim3(EDIM / 64, M / 128), 256, 0, stream>>>(
      FFHbf, W2, b2, Ob, nullptr, M, EDIM, DFF, 0);

  // 9) LN2(x + ff) -> d_out
  ln_kernel<<<M / 8, 256, 0, stream>>>(Xb, Ob, ln2_w, ln2_b, (float*)d_out,
                                       nullptr, M);
}
